// GraphCell_17197049053639
// MI455X (gfx1250) — matrix operand order repaired and checked by reading, not compiled, not run on hardware
//
#include <hip/hip_runtime.h>
#include <hip/hip_bf16.h>

typedef unsigned short u16;
typedef unsigned int   u32;

typedef __attribute__((ext_vector_type(16))) __bf16 bf16x16;
typedef __attribute__((ext_vector_type(8)))  float  f32x8;

#define BM 128
#define BN 128
#define BK 32

static const int B_IMG  = 128;
static const int NOBJ   = 36;
static const int NNODE  = B_IMG * NOBJ;      // 4608
static const float NEG_SLOPE = 0.2f;

__device__ __forceinline__ u16 f32_to_bf16(float f) {
    u32 u = __builtin_bit_cast(u32, f);
    u32 rnd = 0x7fffu + ((u >> 16) & 1u);    // round-to-nearest-even
    u += rnd;
    return (u16)(u >> 16);
}

// LDS byte offset of a shared-memory pointer: flat shared-aperture addresses
// carry the LDS offset in addr[31:0] (ISA flat aperture mapping).
__device__ __forceinline__ u32 lds_off(const void* p) {
    return (u32)(uintptr_t)p;
}

// CDNA5 async copy: 16 bytes per lane, global -> LDS, tracked by ASYNCcnt.
__device__ __forceinline__ void async_copy_b128(u32 lds_byte, const char* gptr) {
    asm volatile("global_load_async_to_lds_b128 %0, %1, off"
                 :: "v"(lds_byte), "v"(gptr) : "memory");
}
__device__ __forceinline__ void wait_async0() {
    asm volatile("s_wait_asynccnt 0" ::: "memory");
}

// ---------------------------------------------------------------------------
// f32 -> bf16 bulk conversion (grid-stride) for activations
// ---------------------------------------------------------------------------
__global__ void cvt_f32_bf16(const float* __restrict__ in, u16* __restrict__ out,
                             size_t n) {
    size_t stride = (size_t)gridDim.x * blockDim.x;
    for (size_t i = (size_t)blockIdx.x * blockDim.x + threadIdx.x; i < n; i += stride)
        out[i] = f32_to_bf16(in[i]);
}

// ---------------------------------------------------------------------------
// Weight transform (one-time): W[K,N] f32 -> Wt[N,K] bf16, 32x32 LDS tiles.
// Transposed weights make every GEMM B-tile row contiguous (async-friendly).
// ---------------------------------------------------------------------------
__global__ __launch_bounds__(256) void cvt_transpose_bf16(
    const float* __restrict__ in, u16* __restrict__ out, int K, int N) {
    __shared__ float tile[32][33];
    int k0 = blockIdx.y * 32, n0 = blockIdx.x * 32;
    int tx = threadIdx.x & 31, ty = threadIdx.x >> 5;      // ty in 0..7
    #pragma unroll
    for (int r = ty; r < 32; r += 8)
        tile[r][tx] = in[(size_t)(k0 + r) * N + n0 + tx];
    __syncthreads();
    #pragma unroll
    for (int r = ty; r < 32; r += 8)
        out[(size_t)(n0 + r) * K + k0 + tx] = f32_to_bf16(tile[tx][r]);
}

// ---------------------------------------------------------------------------
// WMMA bf16 GEMM:  C[M,N] = A[M,K] * Wt[N,K]^T  (+ bias) (* qmul broadcast)
//   A row-major bf16, Wt transposed-weights bf16 (row n is k-contiguous).
//   Block = 256 threads (8 waves), tile 128x128, K-step 32.
//   Waves arranged 4(m) x 2(n): each wave owns 32 rows x 64 cols
//   -> 2 A-frags, 4 B-frags, 8 WMMAs per K-step (B-frags reused twice).
//   Double-buffered LDS staged with global_load_async_to_lds_b128 (ASYNCcnt).
//   M % 128 == 0, N % 128 == 0, K % 32 == 0 (true for all calls here).
// ---------------------------------------------------------------------------
__global__ __launch_bounds__(256) void wmma_gemm_bf16(
    const u16* __restrict__ A,        // [M,K] bf16
    const u16* __restrict__ Bt,       // [N,K] bf16 (transposed weights)
    const float* __restrict__ bias,   // [N] or null
    const float* __restrict__ qmul,   // [M/36, N] or null: out *= qmul[row/36, n]
    float* __restrict__ Cf,           // [M,N] or null
    u16*   __restrict__ Cbf,          // [M,N] or null
    int M, int N, int K)
{
    // double buffers: A tile 128x32 (8KB), B tile 128x32 n-major/k-contig (8KB)
    __shared__ __align__(16) u16 As2[2][BM * BK];
    __shared__ __align__(16) u16 Bs2[2][BN * BK];

    const int tid  = threadIdx.x;
    const int wave = tid >> 5;
    const int lane = tid & 31;
    const int m0 = blockIdx.y * BM;
    const int n0 = blockIdx.x * BN;
    const int mrow = (wave >> 1) * 32;       // wave's rows within tile
    const int ncol = (wave & 1) * 64;        // wave's cols within tile
    const size_t rowbytes = (size_t)K * 2;   // both A and Bt have K-major rows

    f32x8 acc[2][4];
    #pragma unroll
    for (int r = 0; r < 2; ++r)
        #pragma unroll
        for (int t = 0; t < 4; ++t)
            #pragma unroll
            for (int v = 0; v < 8; ++v) acc[r][t][v] = 0.0f;

    // async-stage one K-tile (kk = k offset in elements) into buffer `buf`
    auto stage = [&](int buf, int kk) {
        const char* Ab = (const char*)A  + (size_t)kk * 2;
        const char* Bb = (const char*)Bt + (size_t)kk * 2;
        u32 asb = lds_off(&As2[buf][0]);
        u32 bsb = lds_off(&Bs2[buf][0]);
        // A tile: 128 rows x 64B = 512 x 16B chunks, 2 per thread
        #pragma unroll
        for (int it = 0; it < 2; ++it) {
            int c   = tid + it * 256;
            int row = c >> 2;
            int seg = (c & 3) << 4;
            async_copy_b128(asb + row * 64 + seg,
                            Ab + (size_t)(m0 + row) * rowbytes + seg);
        }
        // B tile: 128 rows x 64B = 512 x 16B chunks, 2 per thread
        #pragma unroll
        for (int it = 0; it < 2; ++it) {
            int c   = tid + it * 256;
            int n   = c >> 2;
            int seg = (c & 3) << 4;
            async_copy_b128(bsb + n * 64 + seg,
                            Bb + (size_t)(n0 + n) * rowbytes + seg);
        }
    };

    const int nsteps = K / BK;
    stage(0, 0);
    wait_async0();
    __syncthreads();

    for (int s = 0; s < nsteps; ++s) {
        const int cur = s & 1;
        if (s + 1 < nsteps) stage(cur ^ 1, (s + 1) * BK);   // prefetch next tile

        const u16* As = As2[cur];
        const u16* Bs = Bs2[cur];

        // ---- 2 A fragments (16x32 bf16, contiguous k = 16*half + s, same as B) ----
        union frag_t { u32 u[8]; bf16x16 v; };
        frag_t afrag[2];
        #pragma unroll
        for (int r = 0; r < 2; ++r) {
            int m    = mrow + r * 16 + (lane & 15);
            int half = lane >> 4;
            #pragma unroll
            for (int v = 0; v < 8; ++v) {
                int k = half * 16 + v * 2;
                afrag[r].u[v] = ((const u32*)As)[(m * BK + k) >> 1];
            }
        }

        // ---- 4 B fragments, each reused for 2 WMMAs -> 8 WMMAs ----
        #pragma unroll
        for (int t = 0; t < 4; ++t) {
            frag_t bfrag;
            int n  = ncol + t * 16 + (lane & 15);
            int kh = (lane >> 4) * 16;
            #pragma unroll
            for (int v = 0; v < 8; ++v)
                bfrag.u[v] = ((const u32*)Bs)[((n * BK + kh) >> 1) + v];

            #pragma unroll
            for (int r = 0; r < 2; ++r)
                acc[r][t] = __builtin_amdgcn_wmma_f32_16x16x32_bf16(
                    false, afrag[r].v, false, bfrag.v, (short)0, acc[r][t],
                    false, false);
        }

        wait_async0();        // our prefetch into the other buffer has landed
        __syncthreads();      // everyone done reading `cur` (implies dscnt wait)
    }

    // ---- epilogue: D layout = vgpr v, lanes 0-15 -> M=v, lanes 16-31 -> M=v+8
    #pragma unroll
    for (int r = 0; r < 2; ++r) {
        const int mw = m0 + mrow + r * 16 + (lane >> 4) * 8;
        #pragma unroll
        for (int t = 0; t < 4; ++t) {
            #pragma unroll
            for (int v = 0; v < 8; ++v) {
                int mg = mw + v;
                int ng = n0 + ncol + t * 16 + (lane & 15);
                float val = acc[r][t][v];
                if (bias) val += bias[ng];
                if (qmul) val *= qmul[(size_t)(mg / NOBJ) * N + ng];
                size_t o = (size_t)mg * N + ng;
                if (Cf)  Cf[o]  = val;
                if (Cbf) Cbf[o] = f32_to_bf16(val);
            }
        }
    }
}

// ---------------------------------------------------------------------------
// Attention logits: al_s[n,h] = <h[n,h,:], a_s[h,:]>, same for a_d.
// One block per node.
// ---------------------------------------------------------------------------
__global__ __launch_bounds__(256) void attn_logits(
    const float* __restrict__ h,      // [N, H*C]
    const float* __restrict__ a_s,    // [H*C]
    const float* __restrict__ a_d,    // [H*C]
    float* __restrict__ als,          // [N,H]
    float* __restrict__ ald,          // [N,H]
    int H, int C)
{
    __shared__ float red[256];
    int n = blockIdx.x;
    const float* row = h + (size_t)n * H * C;
    for (int hh = 0; hh < H; ++hh) {
        float ss = 0.f, sd = 0.f;
        for (int c = threadIdx.x; c < C; c += 256) {
            float x = row[hh * C + c];
            ss += x * a_s[hh * C + c];
            sd += x * a_d[hh * C + c];
        }
        red[threadIdx.x] = ss; __syncthreads();
        for (int s = 128; s > 0; s >>= 1) {
            if (threadIdx.x < s) red[threadIdx.x] += red[threadIdx.x + s];
            __syncthreads();
        }
        if (threadIdx.x == 0) als[(size_t)n * H + hh] = red[0];
        __syncthreads();
        red[threadIdx.x] = sd; __syncthreads();
        for (int s = 128; s > 0; s >>= 1) {
            if (threadIdx.x < s) red[threadIdx.x] += red[threadIdx.x + s];
            __syncthreads();
        }
        if (threadIdx.x == 0) ald[(size_t)n * H + hh] = red[0];
        __syncthreads();
    }
}

// ---------------------------------------------------------------------------
// Dense per-image GAT aggregation with softmax over sources.
//  MEAN=false: block = (image, head); out[j, h*C+c] = relu?(sum + bias) + res
//  MEAN=true : block = image; out[j,c] = (1/H) * sum_h agg + bias[c]
// ---------------------------------------------------------------------------
template <bool MEAN>
__global__ __launch_bounds__(256) void gat_aggregate(
    const float* __restrict__ h,        // [N, H*C]
    const float* __restrict__ als,      // [N,H]
    const float* __restrict__ ald,      // [N,H]
    const float* __restrict__ bias,     // [H*C] (concat) or [C] (mean)
    const float* __restrict__ residual, // [N,H*C] or null
    float* __restrict__ outf,           // [N,H*C] or [N,C]
    u16*   __restrict__ outbf,          // same shape or null
    int H, int C, int do_relu)
{
    const int NH = MEAN ? 5 : 1;
    __shared__ float s_alpha[NH * NOBJ * NOBJ];
    __shared__ float s_als[NH * NOBJ];
    __shared__ float s_ald[NH * NOBJ];

    const int tid = threadIdx.x;
    const int b   = MEAN ? blockIdx.x : (blockIdx.x / H);
    const int h0  = MEAN ? 0 : (blockIdx.x % H);
    const int nh  = MEAN ? H : 1;
    const int base = b * NOBJ;

    for (int i = tid; i < nh * NOBJ; i += 256) {
        int hh = i / NOBJ, j = i % NOBJ;
        s_als[i] = als[(size_t)(base + j) * H + h0 + hh];
        s_ald[i] = ald[(size_t)(base + j) * H + h0 + hh];
    }
    __syncthreads();

    for (int i = tid; i < nh * NOBJ * NOBJ; i += 256) {
        int hh = i / (NOBJ * NOBJ);
        int r  = i % (NOBJ * NOBJ);
        int src = r / NOBJ, dst = r % NOBJ;
        float e = s_als[hh * NOBJ + src] + s_ald[hh * NOBJ + dst];
        s_alpha[i] = (e > 0.f) ? e : NEG_SLOPE * e;
    }
    __syncthreads();

    for (int i = tid; i < nh * NOBJ; i += 256) {
        int hh = i / NOBJ, dst = i % NOBJ;
        float* col = s_alpha + hh * NOBJ * NOBJ;
        float mx = -1e30f;
        for (int s = 0; s < NOBJ; ++s) mx = fmaxf(mx, col[s * NOBJ + dst]);
        float den = 0.f;
        for (int s = 0; s < NOBJ; ++s) {
            float ex = expf(col[s * NOBJ + dst] - mx);
            col[s * NOBJ + dst] = ex;
            den += ex;
        }
        float inv = 1.0f / (den + 1e-16f);
        for (int s = 0; s < NOBJ; ++s) col[s * NOBJ + dst] *= inv;
    }
    __syncthreads();

    for (int idx = tid; idx < NOBJ * C; idx += 256) {
        int dst = idx / C, c = idx % C;
        if (MEAN) {
            float acc = 0.f;
            for (int hh = 0; hh < nh; ++hh) {
                const float* col = s_alpha + hh * NOBJ * NOBJ;
                float a = 0.f;
                for (int src = 0; src < NOBJ; ++src)
                    a += col[src * NOBJ + dst] *
                         h[(size_t)(base + src) * H * C + (size_t)hh * C + c];
                acc += a;
            }
            float val = acc / (float)H + bias[c];
            outf[(size_t)(base + dst) * C + c] = val;
        } else {
            const float* col = s_alpha;
            float acc = 0.f;
            for (int src = 0; src < NOBJ; ++src)
                acc += col[src * NOBJ + dst] *
                       h[(size_t)(base + src) * H * C + (size_t)h0 * C + c];
            float val = acc + bias[h0 * C + c];
            if (do_relu) val = fmaxf(val, 0.f);
            size_t o = (size_t)(base + dst) * H * C + (size_t)h0 * C + c;
            if (residual) val += residual[o];
            outf[o] = val;
            if (outbf) outbf[o] = f32_to_bf16(val);
        }
    }
}

// ---------------------------------------------------------------------------
extern "C" void kernel_launch(void* const* d_in, const int* in_sizes, int n_in,
                              void* d_out, int out_size, void* d_ws, size_t ws_size,
                              hipStream_t stream) {
    (void)in_sizes; (void)n_in; (void)out_size; (void)ws_size;

    const float* qe  = (const float*)d_in[0];   // [128, 2400]
    const float* obj = (const float*)d_in[1];   // [4608, 2048]
    // d_in[2] = edge_index (fully-connected per image; structure known)
    const float* Wq  = (const float*)d_in[3];   const float* bq = (const float*)d_in[4];
    const float* Wv  = (const float*)d_in[5];   const float* bv = (const float*)d_in[6];
    const float* W1  = (const float*)d_in[7];
    const float* a1s = (const float*)d_in[8];   const float* a1d = (const float*)d_in[9];
    const float* b1  = (const float*)d_in[10];
    const float* W2  = (const float*)d_in[11];
    const float* a2s = (const float*)d_in[12];  const float* a2d = (const float*)d_in[13];
    const float* b2  = (const float*)d_in[14];
    const float* W3  = (const float*)d_in[15];
    const float* a3s = (const float*)d_in[16];  const float* a3d = (const float*)d_in[17];
    const float* b3  = (const float*)d_in[18];

    const int QD = 2400, VD = 2048, D1 = 2048, DG = 1024, D3 = 2560;

    // ---- workspace layout (with reuse) ----
    char* ws = (char*)d_ws;
    size_t off = 0;
    auto alloc = [&](size_t bytes) -> char* {
        char* p = ws + off;
        off += (bytes + 255) & ~(size_t)255;
        return p;
    };
    u16*   wq_bf  = (u16*)  alloc((size_t)QD * D1 * 2);   // [2048,2400] transposed
    u16*   wv_bf  = (u16*)  alloc((size_t)VD * D1 * 2);   // [2048,2048] transposed
    u16*   w1_bf  = (u16*)  alloc((size_t)D1 * DG * 2);   // [1024,2048] transposed
    u16*   w2_bf  = (u16*)  alloc((size_t)DG * DG * 2);   // [1024,1024] transposed
    u16*   w3_bf  = (u16*)  alloc((size_t)DG * D3 * 2);   // [2560,1024] transposed
    u16*   qe_bf  = (u16*)  alloc((size_t)B_IMG * QD * 2);
    float* q_f32  = (float*)alloc((size_t)B_IMG * D1 * 4);
    char*  bufA   =         alloc((size_t)NNODE * D1 * 2);  // obj_bf -> h2_f32
    char*  bufB   =         alloc((size_t)NNODE * D1 * 2);  // x_bf   -> g2_f32
    char*  bufC   =         alloc((size_t)NNODE * DG * 4);  // h1_f32 -> g2_bf
    float* g1_f32 = (float*)alloc((size_t)NNODE * DG * 4);
    u16*   g1_bf  = (u16*)  alloc((size_t)NNODE * DG * 2);
    float* h3_f32 = (float*)alloc((size_t)NNODE * D3 * 4);
    float* al_s   = (float*)alloc((size_t)NNODE * 5 * 4);
    float* al_d   = (float*)alloc((size_t)NNODE * 5 * 4);

    u16*   obj_bf = (u16*)bufA;
    float* h2_f32 = (float*)bufA;
    u16*   x_bf   = (u16*)bufB;
    float* g2_f32 = (float*)bufB;
    float* h1_f32 = (float*)bufC;
    u16*   g2_bf  = (u16*)bufC;

    dim3 blk(256);

    // ---- 1) weights: f32 [K,N] -> bf16 [N,K] (one-time transpose) ----
    cvt_transpose_bf16<<<dim3(D1 / 32, QD / 32), blk, 0, stream>>>(Wq, wq_bf, QD, D1);
    cvt_transpose_bf16<<<dim3(D1 / 32, VD / 32), blk, 0, stream>>>(Wv, wv_bf, VD, D1);
    cvt_transpose_bf16<<<dim3(DG / 32, D1 / 32), blk, 0, stream>>>(W1, w1_bf, D1, DG);
    cvt_transpose_bf16<<<dim3(DG / 32, DG / 32), blk, 0, stream>>>(W2, w2_bf, DG, DG);
    cvt_transpose_bf16<<<dim3(D3 / 32, DG / 32), blk, 0, stream>>>(W3, w3_bf, DG, D3);
    // activations: plain f32 -> bf16
    cvt_f32_bf16<<<1024, blk, 0, stream>>>(qe,  qe_bf,  (size_t)B_IMG * QD);
    cvt_f32_bf16<<<2048, blk, 0, stream>>>(obj, obj_bf, (size_t)NNODE * VD);

    // ---- 2) q = qe @ Wq + bq            [128, 2048] fp32 ----
    wmma_gemm_bf16<<<dim3(D1 / BN, B_IMG / BM), blk, 0, stream>>>(
        qe_bf, wq_bf, bq, nullptr, q_f32, nullptr, B_IMG, D1, QD);

    // ---- 3) x = (obj @ Wv + bv) * q[row/36]   -> bf16 only ----
    wmma_gemm_bf16<<<dim3(D1 / BN, NNODE / BM), blk, 0, stream>>>(
        obj_bf, wv_bf, bv, q_f32, nullptr, x_bf, NNODE, D1, VD);

    // ---- 4) layer 1: h1 = x @ W1 ; logits ; aggregate (relu, +b1) ----
    wmma_gemm_bf16<<<dim3(DG / BN, NNODE / BM), blk, 0, stream>>>(
        x_bf, w1_bf, nullptr, nullptr, h1_f32, nullptr, NNODE, DG, D1);
    attn_logits<<<NNODE, blk, 0, stream>>>(h1_f32, a1s, a1d, al_s, al_d, 4, 256);
    gat_aggregate<false><<<B_IMG * 4, blk, 0, stream>>>(
        h1_f32, al_s, al_d, b1, nullptr, g1_f32, g1_bf, 4, 256, 1);

    // ---- 5) layer 2: h2 = g1 @ W2 ; aggregate (relu, +b2) + residual g1 ----
    wmma_gemm_bf16<<<dim3(DG / BN, NNODE / BM), blk, 0, stream>>>(
        g1_bf, w2_bf, nullptr, nullptr, h2_f32, nullptr, NNODE, DG, DG);
    attn_logits<<<NNODE, blk, 0, stream>>>(h2_f32, a2s, a2d, al_s, al_d, 4, 256);
    gat_aggregate<false><<<B_IMG * 4, blk, 0, stream>>>(
        h2_f32, al_s, al_d, b2, g1_f32, g2_f32, g2_bf, 4, 256, 1);

    // ---- 6) layer 3: h3 = g2 @ W3 ; aggregate mean over 5 heads, +b3 ----
    wmma_gemm_bf16<<<dim3(D3 / BN, NNODE / BM), blk, 0, stream>>>(
        g2_bf, w3_bf, nullptr, nullptr, h3_f32, nullptr, NNODE, D3, DG);
    attn_logits<<<NNODE, blk, 0, stream>>>(h3_f32, a3s, a3d, al_s, al_d, 5, 512);
    gat_aggregate<true><<<B_IMG, blk, 0, stream>>>(
        h3_f32, al_s, al_d, b3, nullptr, (float*)d_out, nullptr, 5, 512, 0);
}
